// AdaptiveWaveletTransform_78211354460517
// MI455X (gfx1250) — matrix operand order repaired and checked by reading, not compiled, not run on hardware
//
#include <hip/hip_runtime.h>

// ---------------------------------------------------------------------------
// Adaptive wavelet transform == 8 causal FIR convs (L = 64..2048) over
// [B=16, S=4096, F=64] f32, weighted + |x|>1e-4 masked.
// v_wmma_f32_16x16x32_f16: M=feature(16), N=time(16), K=time-lag.
// A = signal tile (LDS f16, double-buffered, ds_load_b128-fusable layout),
// B = Toeplitz tile read as aligned consecutive dwords (offsets 0..31 from a
// single biased base) from a reversed dilated kernel (2 parity copies, zero
// guard bands) -> no packing VALU, no negative-offset address math,
// no branches around WMMA.
// ---------------------------------------------------------------------------

typedef __attribute__((ext_vector_type(16))) _Float16 v16h;
typedef __attribute__((ext_vector_type(8)))  float    v8f;

#define B_      16
#define S_      4096
#define F_      64
#define NS_     8
#define TT_     64      // output time steps per workgroup
#define KB_     64      // K (time-lag) steps staged per iteration
#define THREADS 256     // 8 wave32s

// L[i] = int(64 * 32^(i/7))  (matches numpy logspace + int())
__constant__ int c_L[NS_] = {64, 105, 172, 282, 463, 760, 1248, 2048};

union FragU { v16h h; unsigned int u[8]; };

__global__ void __launch_bounds__(THREADS)
wavelet_wmma_kernel(const float* __restrict__ sig,
                    const float* __restrict__ mw,    // [8][64] mother wavelets
                    const float* __restrict__ swts,  // [8] scale weights
                    float* __restrict__ out)         // [16][8][4096][64]
{
    // reversed dilated kernel: rev[c][j] = g((M+c) - j), g()==0 outside [0,L)
    __shared__ unsigned int rev_u[2][1122];          // 2 parity copies, dword view
    __shared__ unsigned int sig_u[2][F_ * (KB_/2)];  // [f][u/2] f16x2, double buffered
    __shared__ float        out_s[TT_ * F_];         // swizzled f32 staging

    const int tid = threadIdx.x;
    const int bx  = blockIdx.x;
    const int sc  = bx >> 10;          // 8 scales
    const int b   = (bx >> 6) & 15;    // 16 batches
    const int T   = (bx & 63) << 6;    // 64 time blocks of 64

    const int   L     = c_L[sc];
    const float scale = exp2f((float)sc * (5.0f / 7.0f));   // 32^(sc/7)
    const float amp   = swts[sc] / sqrtf(scale);            // fold weight into kernel
    const float step  = 63.0f / (float)(L - 1);

    // ---- build reversed dilated kernel, both parity copies ----
    const int M    = (L + 127) & ~1;   // even, >= max displacement + 1
    const int NREV = M + 66;           // halfs used per copy (<= 2244)
#pragma unroll
    for (int c = 0; c < 2; ++c) {
        _Float16* rh = (_Float16*)&rev_u[c][0];
        for (int j = tid; j < NREV; j += THREADS) {
            int   d   = M + c - j;
            float val = 0.0f;
            if ((unsigned)d < (unsigned)L) {
                float x  = (float)d * step;
                int   i0 = (int)x; if (i0 > 62) i0 = 62;
                float fr = x - (float)i0;
                float w0 = mw[sc * 64 + i0];
                float w1 = mw[sc * 64 + i0 + 1];
                val = (w0 + (w1 - w0) * fr) * amp;
            }
            rh[j] = (_Float16)val;
        }
    }

    // ---- K-loop setup: u over [T-L+1, T+63] in steps of 64 ----
    const int U_lo = ((T - L + 1) >> 6) << 6;        // arithmetic shift = floor
    const int nblk = (T + TT_ - U_lo) >> 6;          // last U == T

    const int lane = tid & 31;
    const int wv   = tid >> 5;
    const int half = lane >> 4;        // K half: k = 16*half + s (contiguous)
    const int ln   = lane & 15;        // A: M row (feature) / B,D: N col (time)
    const int f0   = (wv & 3) << 4;    // wave's 16-feature tile
    const int Tn0  = T + ((wv >> 2) << 5);  // wave's first 16-time tile (2nd = +16)

    const int par  = ln & 1;
    const int jME  = M + par;          // even + parity(d) => even numerator below
    // Biased B base: fragment dwords are rvp[0..31] (all non-negative offsets):
    //   b10 (tile1,K0) = rvp[0..7],  b00 (tile0,K0) = rvp[8..15],
    //   b11 (tile1,K1) = rvp[16..23], b01 (tile0,K1) = rvp[24..31]
    const unsigned int* rvp =
        &rev_u[par][0] + (((jME - (Tn0 + ln - U_lo - (half << 4))) >> 1) - 8);
    // A-frag dword base: idx = f*32 + k/2, k = half*16 + 2*v
    const int abase = (f0 + ln) * (KB_/2) + half * 8;

    const int    fcol = tid & 63;
    const int    u0p  = (tid >> 6) << 1;             // 0,2,4,6
    const float* gcol = sig + (size_t)b * S_ * F_ + fcol;

    // stage 64 u-steps x 64 features as packed f16x2 (branch-free lanes)
    auto stage = [&](int buf, int U) {
        if (U + 2 * KB_ <= S_)         // uniform scalar guard
            __builtin_prefetch(&gcol[(size_t)(U + 2 * KB_ + u0p) * F_], 0, 3);
        unsigned int* dst = &sig_u[buf][fcol * (KB_/2)];
        if (U >= 0) {                  // uniform: fast path, no clamping
#pragma unroll
            for (int r = 0; r < 8; ++r) {
                int u = u0p + (r << 3);
                union { _Float16 h[2]; unsigned int w; } p;
                p.h[0] = (_Float16)gcol[(size_t)(U + u)     * F_];
                p.h[1] = (_Float16)gcol[(size_t)(U + u + 1) * F_];
                dst[u >> 1] = p.w;
            }
        } else {                       // leading edge: clamp + cndmask, no branches
#pragma unroll
            for (int r = 0; r < 8; ++r) {
                int u  = u0p + (r << 3);
                int g0 = U + u, g1 = g0 + 1;
                float v0 = gcol[(size_t)(g0 < 0 ? 0 : g0) * F_];
                float v1 = gcol[(size_t)(g1 < 0 ? 0 : g1) * F_];
                v0 = (g0 < 0) ? 0.0f : v0;
                v1 = (g1 < 0) ? 0.0f : v1;
                union { _Float16 h[2]; unsigned int w; } p;
                p.h[0] = (_Float16)v0;
                p.h[1] = (_Float16)v1;
                dst[u >> 1] = p.w;
            }
        }
    };

    v8f c0 = {};
    v8f c1 = {};

    stage(0, U_lo);
    __syncthreads();

    for (int kb = 0; kb < nblk; ++kb) {
        const int cur = kb & 1;
        const int U   = U_lo + (kb << 6);
        if (kb + 1 < nblk) stage(cur ^ 1, U + KB_);

        // A: two signal fragments (K 0..31 and 32..63), contiguous k = 16*half + s;
        // each is two 16B-aligned runs of 4 dwords -> ds_load_b128 pairs
        const unsigned int* sg = &sig_u[cur][0];
        FragU a0, a1;
#pragma unroll
        for (int v = 0; v < 8; ++v) {
            int o = v;
            a0.u[v] = sg[abase + o];
            a1.u[v] = sg[abase + 16 + o];
        }

        // B: four Toeplitz fragments = 32 consecutive aligned dwords per lane,
        // all at non-negative immediate offsets from one base
        FragU b00, b10, b01, b11;
#pragma unroll
        for (int v = 0; v < 8; ++v) {
            b10.u[v] = rvp[v];
            b00.u[v] = rvp[8 + v];
            b11.u[v] = rvp[16 + v];
            b01.u[v] = rvp[24 + v];
        }
        rvp += 32;   // U advances by 64 halfs == 32 dwords per iteration

        c0 = __builtin_amdgcn_wmma_f32_16x16x32_f16(false, a0.h, false, b00.h,
                                                    (short)0, c0, false, false);
        c1 = __builtin_amdgcn_wmma_f32_16x16x32_f16(false, a0.h, false, b10.h,
                                                    (short)0, c1, false, false);
        c0 = __builtin_amdgcn_wmma_f32_16x16x32_f16(false, a1.h, false, b01.h,
                                                    (short)0, c0, false, false);
        c1 = __builtin_amdgcn_wmma_f32_16x16x32_f16(false, a1.h, false, b11.h,
                                                    (short)0, c1, false, false);
        __syncthreads();
    }

    // ---- transpose through LDS (swizzled columns) ----
#pragma unroll
    for (int v = 0; v < 8; ++v) {
        int fr  = f0 + (half << 3) + v;   // D layout: VGPR v -> M row (+8 upper half)
        int t0l = (Tn0 - T) + ln;         // D layout: lane&15 -> N col
        out_s[(t0l << 6) + ((fr + t0l) & 63)] = c0[v];
        int t1l = t0l + 16;
        out_s[(t1l << 6) + ((fr + t1l) & 63)] = c1[v];
    }
    __syncthreads();

    // ---- coalesced masked store: 64 rows x 64 features ----
    const int fo = tid & 63;
    const int tb = tid >> 6;
    const size_t obase = (((size_t)b * NS_ + sc) * S_ + T) * F_ + fo;
#pragma unroll
    for (int r = 0; r < 16; ++r) {
        int t = (r << 2) + tb;
        float x = out_s[(t << 6) + ((fo + t) & 63)];
        x = (fabsf(x) > 1e-4f) ? x : 0.0f;   // sparsity threshold (strict >)
        out[obase + (size_t)t * F_] = x;
    }
}

extern "C" void kernel_launch(void* const* d_in, const int* in_sizes, int n_in,
                              void* d_out, int out_size, void* d_ws, size_t ws_size,
                              hipStream_t stream) {
    (void)in_sizes; (void)n_in; (void)out_size; (void)d_ws; (void)ws_size;
    const float* sig = (const float*)d_in[0];
    const float* mw  = (const float*)d_in[1];
    const float* sw  = (const float*)d_in[2];
    float* out = (float*)d_out;

    dim3 grid(NS_ * B_ * (S_ / TT_));   // 8 * 16 * 64 = 8192 blocks
    dim3 block(THREADS);
    wavelet_wmma_kernel<<<grid, block, 0, stream>>>(sig, mw, sw, out);
}
